// GPT_24429773980377
// MI455X (gfx1250) — matrix operand order repaired and checked by reading, not compiled, not run on hardware
//
#include <hip/hip_runtime.h>

// ---------------------------------------------------------------------------
// GPT-2 (L=12, B=2, S=1024, C=768, H=12, V=50257) forward + loss for gfx1250.
// GEMMs on v_wmma_f32_16x16x32_bf16 with double-buffered LDS + register
// prefetch (one barrier per K-tile, global latency hidden under WMMAs).
// Staging uses float2 global loads + packed-bf16x2 b32 LDS stores.
// fp32 -> bf16 conversion happens while staging tiles; fp32 accumulation.
// Deterministic fixed-order reductions only (graph-replay stable).
// ---------------------------------------------------------------------------

typedef __attribute__((ext_vector_type(16))) __bf16 v16bf;
typedef __attribute__((ext_vector_type(8)))  float  v8f;

#define NL_   12
#define NB_   2
#define NS_   1024
#define NC_   768
#define NH_   12
#define ND_   64
#define NFF_  3072
#define NT_   2048      // NB_*NS_ tokens
#define NV_   50257
#define ATT_SCALE 0.125f   // 1/sqrt(64)

static __device__ __forceinline__ unsigned short f32_to_bf16(float f) {
    unsigned int u = __float_as_uint(f);
    u += 0x7FFFu + ((u >> 16) & 1u);          // round-to-nearest-even
    return (unsigned short)(u >> 16);
}

static __device__ __forceinline__ unsigned int pack_bf16x2(float lo, float hi) {
    return (unsigned int)f32_to_bf16(lo) | ((unsigned int)f32_to_bf16(hi) << 16);
}

static __device__ __forceinline__ float gelu_f(float x) {
    const float c0 = 0.7978845608028654f, c1 = 0.044715f;
    return 0.5f * x * (1.0f + tanhf(c0 * (x + c1 * x * x * x)));
}

union WFrag { v16bf v; unsigned int u[8]; };

// ---------------------------------------------------------------------------
// WMMA GEMM:  C[M,N] = act( A[M,K] * B + bias ) (+ res)
//   BT=0: B is [K,N] row-major.  BT=1: B is [N,K] row-major (used for wte^T).
// Block tile 128x128, K-tile 32, double-buffered LDS. 256 threads = 8 waves
// (2x4); each wave owns a 64x32 sub-tile = 4x2 WMMA 16x16x32 fragments.
// Pipeline per K-tile:  global loads (next) -> 8 WMMAs (cur) -> cvt+ds_store
// (next) -> barrier.  M,K are multiples of 128/32; only N needs guards.
// ---------------------------------------------------------------------------
template<bool BT, bool BIAS, bool GELU_, bool RES>
__global__ __launch_bounds__(256)
void gemm_bf16_wmma(const float* __restrict__ A, const float* __restrict__ Bm,
                    const float* __restrict__ bias, const float* res,
                    float* Cp, int M, int N, int K)
{
    __shared__ unsigned short As[2][128][34];  // [buf][m][k], +2 pad
    __shared__ unsigned short Bs[2][128][34];  // [buf][n][k] -> packed u32 frags

    const int tid  = threadIdx.x;
    const int bm   = blockIdx.y, bn = blockIdx.x;
    const int wid  = tid >> 5,  lane = tid & 31;
    const int wm   = wid >> 2,  wn   = wid & 3;
    const int half = lane >> 4, l15  = lane & 15;

    // staging coordinates (fixed per thread)
    const int ak2 = (tid & 15) * 2, am0 = tid >> 4;   // A / BT-B: float2 on k
    const int bnn = tid & 127, bkp = (tid >> 7) * 2;  // B(KN): coalesced on n
    const bool bok = (bn * 128 + bnn) < N;

    v8f acc[4][2];
    #pragma unroll
    for (int i = 0; i < 4; ++i)
        #pragma unroll
        for (int j = 0; j < 2; ++j)
            #pragma unroll
            for (int q = 0; q < 8; ++q) acc[i][j][q] = 0.0f;

    auto load_a = [&](float2* r, int k0) {
        const float* Ab = A + (size_t)(bm * 128 + am0) * K + k0 + ak2;
        #pragma unroll
        for (int i = 0; i < 8; ++i)
            r[i] = *(const float2*)(Ab + (size_t)(i * 16) * K);
    };
    auto store_a = [&](int buf, const float2* r) {
        #pragma unroll
        for (int i = 0; i < 8; ++i)
            *(unsigned int*)&As[buf][am0 + i * 16][ak2] =
                pack_bf16x2(r[i].x, r[i].y);
    };
    auto load_b = [&](float2* r, int k0) {
        if (!BT) {
            const float* Bb = Bm + (size_t)(k0 + bkp) * N + bn * 128 + bnn;
            #pragma unroll
            for (int i = 0; i < 8; ++i) {
                float lo = bok ? Bb[(size_t)(i * 4) * N]     : 0.0f;
                float hi = bok ? Bb[(size_t)(i * 4 + 1) * N] : 0.0f;
                r[i] = make_float2(lo, hi);
            }
        } else {
            #pragma unroll
            for (int i = 0; i < 8; ++i) {
                int gn = bn * 128 + am0 + i * 16;
                r[i] = (gn < N)
                     ? *(const float2*)(Bm + (size_t)gn * K + k0 + ak2)
                     : make_float2(0.0f, 0.0f);
            }
        }
    };
    auto store_b = [&](int buf, const float2* r) {
        if (!BT) {
            #pragma unroll
            for (int i = 0; i < 8; ++i)
                *(unsigned int*)&Bs[buf][bnn][bkp + i * 4] =
                    pack_bf16x2(r[i].x, r[i].y);
        } else {
            #pragma unroll
            for (int i = 0; i < 8; ++i)
                *(unsigned int*)&Bs[buf][am0 + i * 16][ak2] =
                    pack_bf16x2(r[i].x, r[i].y);
        }
    };
    auto compute = [&](int buf) {
        WFrag a[4], b[2];
        #pragma unroll
        for (int i = 0; i < 4; ++i) {
            int row = wm * 64 + i * 16 + l15;
            #pragma unroll
            for (int j = 0; j < 8; ++j) {
                int kb = ((j & 4) ? 16 : 0) + half * 8 + (j & 3) * 2;
                a[i].u[j] = *(const unsigned int*)&As[buf][row][kb];
            }
        }
        #pragma unroll
        for (int j2 = 0; j2 < 2; ++j2) {
            int col = wn * 32 + j2 * 16 + l15;
            #pragma unroll
            for (int j = 0; j < 8; ++j) {
                int kk2 = ((j & 4) ? 16 : 0) + half * 8 + (j & 3) * 2;
                b[j2].u[j] = *(const unsigned int*)&Bs[buf][col][kk2];
            }
        }
        #pragma unroll
        for (int i = 0; i < 4; ++i)
            #pragma unroll
            for (int j = 0; j < 2; ++j)
                acc[i][j] = __builtin_amdgcn_wmma_f32_16x16x32_bf16(
                    false, a[i].v, false, b[j].v, (short)0, acc[i][j],
                    false, false);
    };

    // ---- software pipeline: prologue stages tile 0 ----
    float2 ar[8], br[8];
    load_a(ar, 0);
    load_b(br, 0);
    store_a(0, ar);
    store_b(0, br);
    __syncthreads();

    const int nk = K >> 5;
    for (int kt = 0; kt < nk; ++kt) {
        const int  cur  = kt & 1;
        const bool more = (kt + 1 < nk);
        if (more) {                       // issue next tile's global loads
            load_a(ar, (kt + 1) << 5);
            load_b(br, (kt + 1) << 5);
        }
        compute(cur);                     // 8 WMMAs hide the load latency
        if (more) {                       // convert + stage into other buffer
            store_a(cur ^ 1, ar);
            store_b(cur ^ 1, br);
        }
        __syncthreads();
    }

    // ---- epilogue: bias / gelu / residual, guarded on N ----
    #pragma unroll
    for (int i = 0; i < 4; ++i) {
        #pragma unroll
        for (int j = 0; j < 2; ++j) {
            int col = bn * 128 + wn * 32 + j * 16 + l15;
            if (col < N) {
                int rbase = bm * 128 + wm * 64 + i * 16 + half * 8;
                #pragma unroll
                for (int p = 0; p < 8; ++p) {
                    int r = rbase + p;
                    float v = acc[i][j][p];
                    if (BIAS)  v += bias[col];
                    if (GELU_) v  = gelu_f(v);
                    if (RES)   v += res[(size_t)r * N + col];
                    Cp[(size_t)r * N + col] = v;
                }
            }
        }
    }
}

// ---------------------------------------------------------------------------
// Embedding: h[t,c] = wte[idx[t],c] + wpe[t%S,c]
// ---------------------------------------------------------------------------
__global__ __launch_bounds__(256)
void embed_kernel(const int* __restrict__ idx, const float* __restrict__ wte,
                  const float* __restrict__ wpe, float* __restrict__ h)
{
    int i = blockIdx.x * 256 + threadIdx.x;
    if (i < NT_ * NC_) {
        int t = i / NC_, c = i - t * NC_;
        int s = t & (NS_ - 1);
        h[i] = wte[(size_t)idx[t] * NC_ + c] + wpe[(size_t)s * NC_ + c];
    }
}

// ---------------------------------------------------------------------------
// Deterministic block reductions (256 threads)
// ---------------------------------------------------------------------------
static __device__ __forceinline__ float block_sum256(float v, float* sbuf) {
    int tid = threadIdx.x;
    #pragma unroll
    for (int off = 16; off; off >>= 1) v += __shfl_xor(v, off, 32);
    __syncthreads();
    if ((tid & 31) == 0) sbuf[tid >> 5] = v;
    __syncthreads();
    if (tid == 0) {
        float r = 0.0f;
        #pragma unroll
        for (int i = 0; i < 8; ++i) r += sbuf[i];
        sbuf[0] = r;
    }
    __syncthreads();
    float r = sbuf[0];
    __syncthreads();
    return r;
}

static __device__ __forceinline__ float block_max256(float v, float* sbuf) {
    int tid = threadIdx.x;
    #pragma unroll
    for (int off = 16; off; off >>= 1) v = fmaxf(v, __shfl_xor(v, off, 32));
    __syncthreads();
    if ((tid & 31) == 0) sbuf[tid >> 5] = v;
    __syncthreads();
    if (tid == 0) {
        float r = sbuf[0];
        #pragma unroll
        for (int i = 1; i < 8; ++i) r = fmaxf(r, sbuf[i]);
        sbuf[0] = r;
    }
    __syncthreads();
    float r = sbuf[0];
    __syncthreads();
    return r;
}

// ---------------------------------------------------------------------------
// LayerNorm over C=768, one block per token row
// ---------------------------------------------------------------------------
__global__ __launch_bounds__(256)
void ln_kernel(const float* __restrict__ x, const float* __restrict__ w,
               const float* __restrict__ b, float* __restrict__ y)
{
    __shared__ float sbuf[8];
    size_t base = (size_t)blockIdx.x * NC_;
    float s = 0.0f;
    for (int c = threadIdx.x; c < NC_; c += 256) s += x[base + c];
    float mean = block_sum256(s, sbuf) * (1.0f / NC_);
    float vs = 0.0f;
    for (int c = threadIdx.x; c < NC_; c += 256) {
        float d = x[base + c] - mean; vs += d * d;
    }
    float var = block_sum256(vs, sbuf) * (1.0f / NC_);
    float inv = rsqrtf(var + 1e-5f);
    for (int c = threadIdx.x; c < NC_; c += 256)
        y[base + c] = (x[base + c] - mean) * inv * w[c] + b[c];
}

// ---------------------------------------------------------------------------
// Causal attention. Grid = B*H*(S/8); 8 waves/block, one query row per wave.
// Two-pass softmax with the score row staged in LDS. q.k dot uses float4
// loads; the v accumulation assigns dims (2*lane, 2*lane+1) per lane so both
// loads and the final store are coalesced b64 ops.
// ---------------------------------------------------------------------------
__global__ __launch_bounds__(256)
void attention_kernel(const float* __restrict__ qkv, float* __restrict__ o)
{
    __shared__ __align__(16) float sq[8][ND_];
    __shared__ float sp[8][NS_];

    const int tid  = threadIdx.x;
    const int w    = tid >> 5, lane = tid & 31;
    const int bh   = blockIdx.x / (NS_ / 8);
    const int s0   = (blockIdx.x % (NS_ / 8)) * 8;
    const int b    = bh / NH_, h = bh % NH_;
    const int s    = s0 + w;
    const size_t rowq = ((size_t)(b * NS_ + s)) * (3 * NC_) + h * ND_;

    sq[w][lane]      = qkv[rowq + lane];
    sq[w][lane + 32] = qkv[rowq + lane + 32];
    __syncthreads();

    // pass 1: scores + running max (lane-strided over keys)
    const float4* sq4 = (const float4*)sq[w];
    float lmax = -3.4e38f;
    for (int t = lane; t <= s; t += 32) {
        const float4* kr4 = (const float4*)(qkv +
            ((size_t)(b * NS_ + t)) * (3 * NC_) + NC_ + h * ND_);
        float dot = 0.0f;
        #pragma unroll
        for (int d4 = 0; d4 < ND_ / 4; ++d4) {
            float4 qa = sq4[d4], kb = kr4[d4];
            dot += qa.x * kb.x + qa.y * kb.y + qa.z * kb.z + qa.w * kb.w;
        }
        dot *= ATT_SCALE;
        sp[w][t] = dot;
        lmax = fmaxf(lmax, dot);
    }
    #pragma unroll
    for (int off = 16; off; off >>= 1) lmax = fmaxf(lmax, __shfl_xor(lmax, off, 32));

    float lsum = 0.0f;
    for (int t = lane; t <= s; t += 32) {
        float e = __expf(sp[w][t] - lmax);
        sp[w][t] = e;
        lsum += e;
    }
    #pragma unroll
    for (int off = 16; off; off >>= 1) lsum += __shfl_xor(lsum, off, 32);
    float inv = 1.0f / lsum;
    __syncthreads();

    // pass 2: o[d] = sum_t p[t] * v[t,d]; lane owns d = 2*lane, 2*lane+1
    float o0 = 0.0f, o1 = 0.0f;
    for (int t = 0; t <= s; ++t) {
        const float2* vr2 = (const float2*)(qkv +
            ((size_t)(b * NS_ + t)) * (3 * NC_) + 2 * NC_ + h * ND_);
        float p = sp[w][t];
        float2 v2 = vr2[lane];
        o0 += p * v2.x;
        o1 += p * v2.y;
    }
    float2* orow = (float2*)(o + ((size_t)(b * NS_ + s)) * NC_ + h * ND_);
    orow[lane] = make_float2(o0 * inv, o1 * inv);
}

// ---------------------------------------------------------------------------
// Per-token cross-entropy: tok_loss[t] = logsumexp(logits[t,:]) - logits[t,tgt]
// ---------------------------------------------------------------------------
__global__ __launch_bounds__(256)
void loss_token_kernel(const float* __restrict__ logits,
                       const int* __restrict__ targets,
                       float* __restrict__ tok_loss)
{
    __shared__ float sbuf[8];
    int t = blockIdx.x;
    const float* lg = logits + (size_t)t * NV_;
    float m = -3.4e38f;
    for (int c = threadIdx.x; c < NV_; c += 256) m = fmaxf(m, lg[c]);
    m = block_max256(m, sbuf);
    float s = 0.0f;
    for (int c = threadIdx.x; c < NV_; c += 256) s += __expf(lg[c] - m);
    s = block_sum256(s, sbuf);
    if (threadIdx.x == 0)
        tok_loss[t] = (logf(s) + m) - lg[targets[t]];
}

__global__ __launch_bounds__(256)
void loss_reduce_kernel(const float* __restrict__ tok_loss, float* __restrict__ out)
{
    __shared__ float sbuf[8];
    float s = 0.0f;
    for (int i = threadIdx.x; i < NT_; i += 256) s += tok_loss[i];
    s = block_sum256(s, sbuf);
    if (threadIdx.x == 0) *out = s * (1.0f / NT_);
}

// ---------------------------------------------------------------------------
// Host launcher
// ---------------------------------------------------------------------------
extern "C" void kernel_launch(void* const* d_in, const int* in_sizes, int n_in,
                              void* d_out, int out_size, void* d_ws, size_t ws_size,
                              hipStream_t stream)
{
    (void)in_sizes; (void)n_in; (void)out_size; (void)ws_size;

    const int*   idx     = (const int*)  d_in[0];
    const int*   targets = (const int*)  d_in[1];
    const float* wte     = (const float*)d_in[2];
    const float* wpe     = (const float*)d_in[3];
    const float* ln1_w   = (const float*)d_in[4];
    const float* ln1_b   = (const float*)d_in[5];
    const float* qkv_w   = (const float*)d_in[6];
    const float* qkv_b   = (const float*)d_in[7];
    const float* proj_w  = (const float*)d_in[8];
    const float* proj_b  = (const float*)d_in[9];
    const float* ln2_w   = (const float*)d_in[10];
    const float* ln2_b   = (const float*)d_in[11];
    const float* fc_w    = (const float*)d_in[12];
    const float* fc_b    = (const float*)d_in[13];
    const float* fc2_w   = (const float*)d_in[14];
    const float* fc2_b   = (const float*)d_in[15];
    const float* lnf_w   = (const float*)d_in[16];
    const float* lnf_b   = (const float*)d_in[17];

    float* logits = (float*)d_out;                       // [T, V] fp32
    float* loss   = logits + (size_t)NT_ * NV_;          // scalar

    char* p = (char*)d_ws;
    float* h    = (float*)p; p += (size_t)NT_ * NC_  * sizeof(float);
    float* x    = (float*)p; p += (size_t)NT_ * NC_  * sizeof(float);
    float* qkv  = (float*)p; p += (size_t)NT_ * 3 * NC_ * sizeof(float);
    float* attn = (float*)p; p += (size_t)NT_ * NC_  * sizeof(float);
    float* ff1  = (float*)p; p += (size_t)NT_ * NFF_ * sizeof(float);
    float* tokl = (float*)p; p += (size_t)NT_ * sizeof(float);

    const dim3 blk(256);

    embed_kernel<<<(NT_ * NC_ + 255) / 256, blk, 0, stream>>>(idx, wte, wpe, h);

    for (int l = 0; l < NL_; ++l) {
        // x = LN1(h)
        ln_kernel<<<NT_, blk, 0, stream>>>(h, ln1_w + l * NC_, ln1_b + l * NC_, x);
        // qkv = x @ qkv_w + qkv_b      [T, 3C]
        gemm_bf16_wmma<false, true, false, false>
            <<<dim3((3 * NC_) / 128, NT_ / 128), blk, 0, stream>>>(
                x, qkv_w + (size_t)l * NC_ * 3 * NC_, qkv_b + (size_t)l * 3 * NC_,
                nullptr, qkv, NT_, 3 * NC_, NC_);
        // attn = softmax(causal(q k^T / sqrt(d))) v
        attention_kernel<<<NB_ * NH_ * (NS_ / 8), blk, 0, stream>>>(qkv, attn);
        // h = h + attn @ proj_w + proj_b
        gemm_bf16_wmma<false, true, false, true>
            <<<dim3(NC_ / 128, NT_ / 128), blk, 0, stream>>>(
                attn, proj_w + (size_t)l * NC_ * NC_, proj_b + (size_t)l * NC_,
                h, h, NT_, NC_, NC_);
        // x = LN2(h)
        ln_kernel<<<NT_, blk, 0, stream>>>(h, ln2_w + l * NC_, ln2_b + l * NC_, x);
        // ff1 = gelu(x @ fc_w + fc_b)   [T, FF]
        gemm_bf16_wmma<false, true, true, false>
            <<<dim3(NFF_ / 128, NT_ / 128), blk, 0, stream>>>(
                x, fc_w + (size_t)l * NC_ * NFF_, fc_b + (size_t)l * NFF_,
                nullptr, ff1, NT_, NFF_, NC_);
        // h = h + ff1 @ fc2_w + fc2_b
        gemm_bf16_wmma<false, true, false, true>
            <<<dim3(NC_ / 128, NT_ / 128), blk, 0, stream>>>(
                ff1, fc2_w + (size_t)l * NFF_ * NC_, fc2_b + (size_t)l * NC_,
                h, h, NT_, NC_, NFF_);
    }

    // x = LNf(h);  logits = x @ wte^T   (BT path, N = V = 50257)
    ln_kernel<<<NT_, blk, 0, stream>>>(h, lnf_w, lnf_b, x);
    gemm_bf16_wmma<true, false, false, false>
        <<<dim3((NV_ + 127) / 128, NT_ / 128), blk, 0, stream>>>(
            x, wte, nullptr, nullptr, logits, NT_, NV_, NC_);

    // loss = mean_t( logsumexp(logits[t]) - logits[t, target[t]] )
    loss_token_kernel<<<NT_, blk, 0, stream>>>(logits, targets, tokl);
    loss_reduce_kernel<<<1, blk, 0, stream>>>(tokl, loss);
}
